// LSTMModulebis_89361089560880
// MI455X (gfx1250) — matrix operand order repaired and checked by reading, not compiled, not run on hardware
//
#include <hip/hip_runtime.h>

// Fused LSTM for MI455X (gfx1250, wave32, WMMA + async-to-LDS pipeline).
//
// One wave per workgroup owns a 16-row batch tile for all T=512 steps.
// Per step: gates[16x48] = x_t[16x64] @ W_ih^T + h[16x12] @ W_hh^T + (b_ih+b_hh)
// via 9x v_wmma_f32_16x16x32_f16, then the LSTM cell elementwise.
//
// x is streamed once from HBM (512 MB) and lstm_out written once (96 MB):
// ~26 us roofline floor at 23.3 TB/s. Because the T-scan is sequential and
// there are only B/16 = 256 waves device-wide (no co-resident waves to hide
// stalls), HBM latency must be hidden *inside* the wave: an 8-slot LDS ring
// is filled 7 steps ahead with GLOBAL_LOAD_ASYNC_TO_LDS_B128 (ASYNCcnt, no
// VGPR cost; 56 ops in flight < 63 counter limit), consumed after
// s_wait_asynccnt 48. Intra-wave LDS ordering uses wavefront-scope fences
// (LDS is in-order within a wave) instead of __syncthreads, so the async
// pipeline is never drained.

typedef __attribute__((ext_vector_type(16))) _Float16 v16h;
typedef __attribute__((ext_vector_type(8)))  float    v8f;
typedef __attribute__((ext_vector_type(4)))  float    f32x4;

namespace {
constexpr int kB = 4096;
constexpr int kT = 512;
constexpr int kI = 64;
constexpr int kH = 12;
constexpr int GP = 52;             // padded gate row stride (floats)
constexpr int kDepth = 8;          // x ring slots
constexpr int kRowF  = 68;         // padded floats per x-tile row in LDS
constexpr int kTileF = 16 * kRowF; // floats per ring slot (4352 B)
}

__device__ __forceinline__ v8f wmma_f16(v16h a, v16h b, v8f c) {
  // D = A(16x32 f16) x B(32x16 f16) + C(16x16 f32)
  return __builtin_amdgcn_wmma_f32_16x16x32_f16(
      /*neg_a=*/false, a, /*neg_b=*/false, b,
      /*c_mod=*/(short)0, c, /*reuse_a=*/false, /*reuse_b=*/false);
}

__device__ __forceinline__ float fast_sigmoid(float v) {
  return __builtin_amdgcn_rcpf(1.0f + __expf(-v));
}
__device__ __forceinline__ float fast_tanh(float v) {
  return 1.0f - 2.0f * __builtin_amdgcn_rcpf(1.0f + __expf(2.0f * v));
}

// Wavefront-scope ordering: LDS ops from one wave execute in order (DScnt),
// so a compiler-level fence is sufficient and drains no counters.
__device__ __forceinline__ void lds_fence() {
  __builtin_amdgcn_fence(__ATOMIC_ACQ_REL, "wavefront");
  __builtin_amdgcn_wave_barrier();
}

// A-operand f16 conversion. 16-bit A 16x32 VGPR layout (ISA 7.12.2):
// elems 0-7 <-> K = kh*8 + e, elems 8-15 <-> K = 16 + kh*8 + e; A1 adds +32.
__device__ __forceinline__ void cvt_x(const f32x4 xf[8], v16h& A0, v16h& A1) {
#pragma unroll
  for (int q = 0; q < 2; ++q) {
#pragma unroll
    for (int e = 0; e < 4; ++e) {
      A0[q * 4 + e]     = (_Float16)xf[q][e];
      A0[8 + q * 4 + e] = (_Float16)xf[2 + q][e];
      A1[q * 4 + e]     = (_Float16)xf[4 + q][e];
      A1[8 + q * 4 + e] = (_Float16)xf[6 + q][e];
    }
  }
}

// Issue one step-group of 8 async 16B copies: instr q moves rows 2q+kh of the
// 16x64 x-tile (cols 4n..4n+3) into the padded LDS ring slot (t & 7).
__device__ __forceinline__ void issue_group(const float* const (&gq)[8],
                                            const unsigned (&lq)[8], int t) {
  const unsigned slotoff = (unsigned)((t & (kDepth - 1)) * (kTileF * 4));
#pragma unroll
  for (int q = 0; q < 8; ++q) {
    const float* g = gq[q] + (size_t)t * kI;
    const unsigned l = lq[q] + slotoff;
    asm volatile("global_load_async_to_lds_b128 %0, %1, off"
                 :: "v"(l), "v"(g)
                 : "memory");
  }
}

__global__ __launch_bounds__(32)
void lstm_fused_kernel(const float* __restrict__ x,     // [B, T, I]
                       const float* __restrict__ Wih,   // [4H, I]
                       const float* __restrict__ Whh,   // [4H, H]
                       const float* __restrict__ bih,   // [4H]
                       const float* __restrict__ bhh,   // [4H]
                       float* __restrict__ out) {       // [B*T*H] ++ hn[B*H] ++ cn[B*H]
  __shared__ float ring[kDepth * kTileF];   // 34816 B async x pipeline
  __shared__ float lds_g[16 * GP];          // gates staging (padded rows)
  __shared__ float lds_h[16 * kH];          // h state, flat p = m*12 + j

  const int lane = threadIdx.x & 31;
  const int n    = lane & 15;   // column (gate col for B/C, batch row for A)
  const int kh   = lane >> 4;   // K-half selector
  const int b0   = blockIdx.x * 16;

  // ---- loop-invariant weight operands in VGPRs ----
  // B 32x16 16-bit layout: elems 0-7 <-> K = kh*8 + e, elems 8-15 <-> K = 16 + kh*8 + e, lane n = N column.
  v16h Bi[3][2];  // B[k,g] = W_ih[g,k]
  v16h Bh[3];     // B[k,g] = W_hh[g,k], K=12 zero-padded to 32
  float biasv[3];
#pragma unroll
  for (int tt = 0; tt < 3; ++tt) {
    const int g = tt * 16 + n;
#pragma unroll
    for (int kc = 0; kc < 2; ++kc) {
      const float* src = Wih + g * kI + kc * 32 + kh * 8;
#pragma unroll
      for (int e = 0; e < 16; ++e) Bi[tt][kc][e] = (_Float16)src[e < 8 ? e : e + 8];
    }
#pragma unroll
    for (int e = 0; e < 16; ++e) {
      const float v = (e < 8 && kh * 8 + e < kH) ? Whh[g * kH + kh * 8 + e] : 0.0f;
      Bh[tt][e] = (_Float16)v;
    }
    biasv[tt] = bih[tt * 16 + n] + bhh[tt * 16 + n];
  }

  // ---- async copy per-lane addressing ----
  const float* gq[8];
  unsigned lq[8];
  const unsigned ring0 = (unsigned)(size_t)&ring[0];
#pragma unroll
  for (int q = 0; q < 8; ++q) {
    const int row = 2 * q + kh;   // batch row within tile
    gq[q] = x + ((size_t)(b0 + row) * kT) * kI + 4 * n;
    lq[q] = ring0 + (unsigned)((row * kRowF + 4 * n) * 4);
  }

  // ---- elementwise lane mapping: 6 (m,j) pairs/lane, p = r*32+lane = m*12+j
  int pm[6], pj[6];
#pragma unroll
  for (int r = 0; r < 6; ++r) {
    const int p = r * 32 + lane;
    pm[r] = p / 12;
    pj[r] = p % 12;
  }

  // ---- state ----
  float creg[6];
#pragma unroll
  for (int r = 0; r < 6; ++r) creg[r] = 0.0f;
  v16h Ah;  // h as 16x32 f16 A operand (zero-padded); h0 = 0
#pragma unroll
  for (int e = 0; e < 16; ++e) Ah[e] = (_Float16)0.0f;

  // ---- prime the async pipeline: groups for t = 0..6 (56 ops in flight) ----
#pragma unroll
  for (int pt = 0; pt < kDepth - 1; ++pt) issue_group(gq, lq, pt);

  v16h A0, A1;
  for (int t = 0; t < kT; ++t) {
    // group t complete when <= 6 groups (48 ops) remain outstanding
    if (t + 7 <= kT) {
      asm volatile("s_wait_asynccnt 48" ::: "memory");
    } else {
      asm volatile("s_wait_asynccnt 0" ::: "memory");
    }

    // ---- read x tile for step t from ring (ds_load_b128, conflict-min) ----
    {
      const float* tp = &ring[(size_t)(t & (kDepth - 1)) * kTileF + n * kRowF + kh * 8];
      f32x4 xf[8];
#pragma unroll
      for (int c = 0; c < 4; ++c) {
        const f32x4* qp = (const f32x4*)(tp + c * 16);
        xf[2 * c]     = qp[0];
        xf[2 * c + 1] = qp[1];
      }
      cvt_x(xf, A0, A1);
    }

    // refill the slot consumed one step ago with step t+7
    if (t + 7 < kT) issue_group(gq, lq, t + 7);

    // ---- gates = bias + x_t @ W_ih^T + h @ W_hh^T  (9 WMMAs) ----
    v8f acc0, acc1, acc2;
#pragma unroll
    for (int e = 0; e < 8; ++e) {
      acc0[e] = biasv[0];
      acc1[e] = biasv[1];
      acc2[e] = biasv[2];
    }
    acc0 = wmma_f16(A0, Bi[0][0], acc0);
    acc1 = wmma_f16(A0, Bi[1][0], acc1);
    acc2 = wmma_f16(A0, Bi[2][0], acc2);
    acc0 = wmma_f16(A1, Bi[0][1], acc0);
    acc1 = wmma_f16(A1, Bi[1][1], acc1);
    acc2 = wmma_f16(A1, Bi[2][1], acc2);
    acc0 = wmma_f16(Ah, Bh[0], acc0);
    acc1 = wmma_f16(Ah, Bh[1], acc1);
    acc2 = wmma_f16(Ah, Bh[2], acc2);

    // ---- scatter C (elem r <-> m = kh*8 + r, col n) into LDS gate tile ----
    {
      const int mbase = kh * 8;
#pragma unroll
      for (int e = 0; e < 8; ++e) {
        float* row = &lds_g[(mbase + e) * GP];
        row[n]      = acc0[e];
        row[16 + n] = acc1[e];
        row[32 + n] = acc2[e];
      }
    }
    lds_fence();

    // ---- LSTM cell elementwise (gate order i, f, g, o) ----
#pragma unroll
    for (int r = 0; r < 6; ++r) {
      const int m = pm[r], j = pj[r];
      const float* gm = &lds_g[m * GP];
      const float iv = fast_sigmoid(gm[j]);
      const float fv = fast_sigmoid(gm[12 + j]);
      const float gv = fast_tanh(gm[24 + j]);
      const float ov = fast_sigmoid(gm[36 + j]);
      const float cnew = fv * creg[r] + iv * gv;
      const float hnew = ov * fast_tanh(cnew);
      creg[r] = cnew;
      lds_h[r * 32 + lane] = hnew;  // p = m*12 + j
      out[((size_t)(b0 + m) * kT + t) * kH + j] = hnew;
    }
    lds_fence();

    // ---- rebuild Ah (16-bit A layout; nonzero K = 0..11) ----
    if (kh == 0) {
#pragma unroll
      for (int e = 0; e < 8; ++e) Ah[e] = (_Float16)lds_h[n * kH + e];        // K=0..7
    } else {
#pragma unroll
      for (int e = 0; e < 4; ++e) Ah[e] = (_Float16)lds_h[n * kH + 8 + e];    // K=8..11
    }
  }

  // ---- hn, cn ([1, B, H] each, concatenated after lstm_out) ----
  float* hn = out + (size_t)kB * kT * kH;
  float* cn = hn + (size_t)kB * kH;
#pragma unroll
  for (int r = 0; r < 6; ++r) {
    const int p = r * 32 + lane;
    hn[(size_t)b0 * kH + p] = lds_h[p];
    cn[(size_t)b0 * kH + p] = creg[r];
  }
}

extern "C" void kernel_launch(void* const* d_in, const int* in_sizes, int n_in,
                              void* d_out, int out_size, void* d_ws, size_t ws_size,
                              hipStream_t stream) {
  (void)in_sizes; (void)n_in; (void)d_ws; (void)ws_size; (void)out_size;
  const float* x   = (const float*)d_in[0];
  const float* Wih = (const float*)d_in[1];
  const float* Whh = (const float*)d_in[2];
  const float* bih = (const float*)d_in[3];
  const float* bhh = (const float*)d_in[4];
  float* out = (float*)d_out;

  dim3 grid(kB / 16);   // 256 single-wave workgroups (one 16-row batch tile each)
  dim3 block(32);
  hipLaunchKernelGGL(lstm_fused_kernel, grid, block, 0, stream,
                     x, Wih, Whh, bih, bhh, out);
}
